// PAConv_32263794328111
// MI455X (gfx1250) — matrix operand order repaired and checked by reading, not compiled, not run on hardware
//
#include <hip/hip_runtime.h>
#include <hip/hip_bf16.h>

#define BATCH 32
#define NPTS  1024
#define KNN   20
#define MSC   8
#define ROWS  (BATCH*NPTS)   /* 32768 */
#define EPSBN 1e-5f

typedef _Float16 f16;
typedef __attribute__((ext_vector_type(16))) _Float16 v16h;
typedef __attribute__((ext_vector_type(8)))  float    v8f;

/* ---------------- x (B,3,N) -> Xt (B*N, 3) ---------------- */
__global__ void paconv_xt_kernel(const float* __restrict__ x, float* __restrict__ Xt) {
    int t = blockIdx.x * blockDim.x + threadIdx.x;
    if (t >= ROWS * 3) return;
    int r = t / 3, c = t % 3;
    int b = r >> 10, n = r & (NPTS - 1);
    Xt[t] = x[(long)b * 3 * NPTS + (long)c * NPTS + n];
}

/* ---------------- kNN: top-20 of neg_dist per point ---------------- */
__global__ void paconv_knn_kernel(const float* __restrict__ x, int* __restrict__ idx) {
    __shared__ float sx0[NPTS], sx1[NPTS], sx2[NPTS], sxx[NPTS];
    const int b = blockIdx.x / (NPTS / 256);
    const int chunk = blockIdx.x % (NPTS / 256);
    const float* xb = x + (long)b * 3 * NPTS;
    for (int i = threadIdx.x; i < NPTS; i += 256) {
        float a0 = xb[i], a1 = xb[NPTS + i], a2 = xb[2 * NPTS + i];
        sx0[i] = a0; sx1[i] = a1; sx2[i] = a2;
        sxx[i] = a0 * a0 + a1 * a1 + a2 * a2;
    }
    __syncthreads();
    const int n = chunk * 256 + threadIdx.x;
    float best[KNN]; int bidx[KNN];
#pragma unroll
    for (int j = 0; j < KNN; ++j) { best[j] = -1e30f; bidx[j] = 0; }
    const float a0 = sx0[n], a1 = sx1[n], a2 = sx2[n], an = sxx[n];
    for (int m = 0; m < NPTS; ++m) {
        float inner = a0 * sx0[m] + a1 * sx1[m] + a2 * sx2[m];
        float nd = 2.f * inner - an - sxx[m];
        if (nd > best[KNN - 1]) {
            int j = KNN - 1;
            while (j > 0 && best[j - 1] < nd) { best[j] = best[j - 1]; bidx[j] = bidx[j - 1]; --j; }
            best[j] = nd; bidx[j] = m;
        }
    }
    int* o = idx + ((long)b * NPTS + n) * KNN;
#pragma unroll
    for (int j = 0; j < KNN; ++j) o[j] = bidx[j];
}

/* ------------- feat_trans weight prep: fp32 (2d,Ncols) -> f16 N-major [Ncols][Kpad] ------------- */
__global__ void paconv_wprep_kernel(const float* __restrict__ mat, int d, int Ncols, int Kpad,
                                    f16* __restrict__ Wp, f16* __restrict__ Wc) {
    int t = blockIdx.x * blockDim.x + threadIdx.x;
    if (t >= Ncols * Kpad) return;
    int n = t / Kpad, k = t % Kpad;
    float c = 0.f, p = 0.f;
    if (k < d) {
        c = mat[(long)k * Ncols + n];
        p = c + mat[(long)(d + k) * Ncols + n];
    }
    Wc[t] = (f16)c;
    Wp[t] = (f16)p;
}

/* conv5_w (1024,512) is already N-major: just convert */
__global__ void paconv_w5prep_kernel(const float* __restrict__ w, f16* __restrict__ Wh, int total) {
    int t = blockIdx.x * blockDim.x + threadIdx.x;
    if (t < total) Wh[t] = (f16)w[t];
}

/* ------------- A pack: fp32 [r][lda] -> fragment-ready f16 [r][Kpad], zero-padded.
   Within each 32-wide K-block, K is stored contiguously (k = 16*half + s)
   so one lane's whole WMMA A-fragment (16 halfs) is a single contiguous 32B load. ------------- */
__global__ void paconv_aprep_kernel(const float* __restrict__ src, int lda, int Kact, int Kpad,
                                    f16* __restrict__ Ah, long total) {
    long t = (long)blockIdx.x * blockDim.x + threadIdx.x;
    if (t >= total) return;
    int r   = (int)(t / Kpad);
    int pos = (int)(t % Kpad);
    int blk = pos >> 5, w = pos & 31, g = w >> 3;
    int k = (blk << 5) + w; (void)g;
    Ah[t] = (k < Kact) ? (f16)src[(long)r * lda + k] : (f16)0.f;
}

/* ---------------- scorenet pass 1: per-hidden-channel sum/sumsq of xyz@w1 ---------------- */
__global__ void paconv_sn_stats_kernel(const float* __restrict__ x, const int* __restrict__ idx,
                                       const float* __restrict__ w1, float* __restrict__ stats) {
    __shared__ float sxyz[KNN][6];
    __shared__ float red[32];
    const int bn = blockIdx.x;
    const int b = bn >> 10, n = bn & (NPTS - 1);
    const float* xb = x + (long)b * 3 * NPTS;
    if (threadIdx.x < KNN) {
        int j = idx[(long)bn * KNN + threadIdx.x];
        float c0 = xb[n], c1 = xb[NPTS + n], c2 = xb[2 * NPTS + n];
        float p0 = xb[j], p1 = xb[NPTS + j], p2 = xb[2 * NPTS + j];
        sxyz[threadIdx.x][0] = p0 - c0; sxyz[threadIdx.x][1] = p1 - c1; sxyz[threadIdx.x][2] = p2 - c2;
        sxyz[threadIdx.x][3] = p0;      sxyz[threadIdx.x][4] = p1;      sxyz[threadIdx.x][5] = p2;
    }
    if (threadIdx.x < 32) red[threadIdx.x] = 0.f;
    __syncthreads();
    const int k = threadIdx.x >> 4, h = threadIdx.x & 15;   /* 320 threads => k<20 */
    float e = 0.f;
#pragma unroll
    for (int c = 0; c < 6; ++c) e += w1[h * 6 + c] * sxyz[k][c];
    atomicAdd(&red[h], e);
    atomicAdd(&red[16 + h], e * e);
    __syncthreads();
    if (threadIdx.x < 32) atomicAdd(&stats[threadIdx.x], red[threadIdx.x]);
}

/* ---------------- scorenet pass 2: bn->relu->fc->softmax+0.5 ---------------- */
__global__ void paconv_sn_score_kernel(const float* __restrict__ x, const int* __restrict__ idx,
                                       const float* __restrict__ w1, const float* __restrict__ w2,
                                       const float* __restrict__ b2, const float* __restrict__ stats,
                                       float* __restrict__ score) {
    __shared__ float sxyz[KNN][6];
    __shared__ float sh[KNN][16];
    __shared__ float ss[KNN][MSC];
    const int bn = blockIdx.x;
    const int b = bn >> 10, n = bn & (NPTS - 1);
    const float* xb = x + (long)b * 3 * NPTS;
    if (threadIdx.x < KNN) {
        int j = idx[(long)bn * KNN + threadIdx.x];
        float c0 = xb[n], c1 = xb[NPTS + n], c2 = xb[2 * NPTS + n];
        float p0 = xb[j], p1 = xb[NPTS + j], p2 = xb[2 * NPTS + j];
        sxyz[threadIdx.x][0] = p0 - c0; sxyz[threadIdx.x][1] = p1 - c1; sxyz[threadIdx.x][2] = p2 - c2;
        sxyz[threadIdx.x][3] = p0;      sxyz[threadIdx.x][4] = p1;      sxyz[threadIdx.x][5] = p2;
    }
    __syncthreads();
    {
        const int k = threadIdx.x >> 4, h = threadIdx.x & 15;
        float e = 0.f;
#pragma unroll
        for (int c = 0; c < 6; ++c) e += w1[h * 6 + c] * sxyz[k][c];
        const float cnt = (float)ROWS * (float)KNN;
        float mu = stats[h] / cnt;
        float var = stats[16 + h] / cnt - mu * mu;
        float v = (e - mu) * rsqrtf(var + EPSBN);
        sh[k][h] = v > 0.f ? v : 0.f;
    }
    __syncthreads();
    if (threadIdx.x < KNN * MSC) {
        const int k = threadIdx.x >> 3, m = threadIdx.x & 7;
        float s = b2[m];
#pragma unroll
        for (int h = 0; h < 16; ++h) s += sh[k][h] * w2[m * 16 + h];
        ss[k][m] = s;
    }
    __syncthreads();
    if (threadIdx.x < KNN * MSC) {
        const int k = threadIdx.x >> 3, m = threadIdx.x & 7;
        float mx = ss[k][0];
#pragma unroll
        for (int j = 1; j < MSC; ++j) mx = fmaxf(mx, ss[k][j]);
        float den = 0.f;
#pragma unroll
        for (int j = 0; j < MSC; ++j) den += __expf(ss[k][j] - mx);
        score[((long)bn * KNN + k) * MSC + m] = __expf(ss[k][m] - mx) / den + 0.5f;
    }
}

/* ---------------- WMMA GEMM: C[M,N] = Ah(f16 packed [M][Kpad]) * Bh(f16 N-major [N][Kpad])
   One wave: 1 M-tile x 4 N-tiles; A fragment loaded once per K-step, reused across 4 WMMAs. -------- */
__global__ __launch_bounds__(256)
void paconv_gemm_wmma_kernel(const f16* __restrict__ Ah, const f16* __restrict__ Bh,
                             int Kpad, float* __restrict__ C, int ldc,
                             int Mrows, int Ncols) {
    const int lane = threadIdx.x & 31;
    const int wave = threadIdx.x >> 5;
    const int tilesN4 = Ncols >> 6;                 /* groups of 4 N-tiles */
    const int numT = (Mrows >> 4) * tilesN4;
    const int t = blockIdx.x * (blockDim.x >> 5) + wave;
    if (t >= numT) return;                          /* wave-uniform: EXEC stays all-ones */
    const int tm  = t / tilesN4;
    const int tn0 = (t % tilesN4) << 2;

    const int kg = lane >> 4;                       /* K-group 0/1 (ISA fragment layout) */
    const f16* arow = Ah + (long)(tm * 16 + (lane & 15)) * Kpad;
    const f16* bcol = Bh + (long)(tn0 * 16 + (lane & 15)) * Kpad;
    const long bstep = (long)16 * Kpad;             /* next N-tile, same lane column */

    v8f acc0 = {}, acc1 = {}, acc2 = {}, acc3 = {};
    for (int kb = 0; kb < Kpad; kb += 32) {
        /* packed layout => whole A fragment is one contiguous 32B load */
        v16h a  = *(const v16h*)(arow + kb + kg * 16);
        v16h b0 = *(const v16h*)(bcol + 0 * bstep + kb + kg * 16);
        v16h b1 = *(const v16h*)(bcol + 1 * bstep + kb + kg * 16);
        v16h b2 = *(const v16h*)(bcol + 2 * bstep + kb + kg * 16);
        v16h b3 = *(const v16h*)(bcol + 3 * bstep + kb + kg * 16);
        if (kb + 32 < Kpad) __builtin_prefetch(arow + kb + 32, 0, 1);
        acc0 = __builtin_amdgcn_wmma_f32_16x16x32_f16(false, a, false, b0, (short)0, acc0, false, false);
        acc1 = __builtin_amdgcn_wmma_f32_16x16x32_f16(false, a, false, b1, (short)0, acc1, false, false);
        acc2 = __builtin_amdgcn_wmma_f32_16x16x32_f16(false, a, false, b2, (short)0, acc2, false, false);
        acc3 = __builtin_amdgcn_wmma_f32_16x16x32_f16(false, a, false, b3, (short)0, acc3, false, false);
    }
    /* C/D layout: reg j -> M = tm*16 + j + 8*kg, N = tile*16 + (lane&15) */
    const int cm = tm * 16 + 8 * kg;
    const int cn = tn0 * 16 + (lane & 15);
#pragma unroll
    for (int j = 0; j < 8; ++j) {
        float* crow = C + (long)(cm + j) * ldc + cn;
        crow[0]  = acc0[j];
        crow[16] = acc1[j];
        crow[32] = acc2[j];
        crow[48] = acc3[j];
    }
}

/* ---------------- assemble: out = sum_k sum_m s*point[idx] - sum_m ssum*center ---------------- */
__global__ void paconv_assemble_kernel(const float* __restrict__ score, const int* __restrict__ idx,
                                       const float* __restrict__ Point, const float* __restrict__ Center,
                                       int o, float* __restrict__ out, int ldo) {
    __shared__ float ssc[KNN][MSC];
    __shared__ float ssum[MSC];
    __shared__ int   sidx[KNN];
    const int bn = blockIdx.x;
    for (int t = threadIdx.x; t < KNN * MSC; t += blockDim.x)
        ssc[t / MSC][t % MSC] = score[(long)bn * KNN * MSC + t];
    if (threadIdx.x < KNN) sidx[threadIdx.x] = idx[(long)bn * KNN + threadIdx.x];
    __syncthreads();
    if (threadIdx.x < MSC) {
        float s = 0.f;
#pragma unroll
        for (int k = 0; k < KNN; ++k) s += ssc[k][threadIdx.x];
        ssum[threadIdx.x] = s;
    }
    __syncthreads();
    const int oid = threadIdx.x;               /* blockDim == o */
    const int Mo = MSC * o;
    const int bbase = (bn >> 10) << 10;        /* b*N */
    const float* crow = Center + (long)bn * Mo + oid;
    float accc = 0.f;
#pragma unroll
    for (int m = 0; m < MSC; ++m) accc += ssum[m] * crow[(long)m * o];
    float accp = 0.f;
    for (int k = 0; k < KNN; ++k) {
        const float* prow = Point + (long)(bbase + sidx[k]) * Mo + oid;
        const float* sk = &ssc[k][0];
#pragma unroll
        for (int m = 0; m < MSC; ++m) accp += sk[m] * prow[(long)m * o];
    }
    out[(long)bn * ldo + oid] = accp - accc;
}

/* ---------------- per-channel sum/sumsq over rows (one block per channel) ---------------- */
__global__ void paconv_bn_stats_kernel(const float* __restrict__ src, int rows, int ldc,
                                       float* __restrict__ sums) {
    __shared__ float s1[256], s2[256];
    const int c = blockIdx.x;
    float a = 0.f, q = 0.f;
    for (int r = threadIdx.x; r < rows; r += 256) {
        float v = src[(long)r * ldc + c];
        a += v; q += v * v;
    }
    s1[threadIdx.x] = a; s2[threadIdx.x] = q;
    __syncthreads();
    for (int st = 128; st > 0; st >>= 1) {
        if (threadIdx.x < st) { s1[threadIdx.x] += s1[threadIdx.x + st]; s2[threadIdx.x] += s2[threadIdx.x + st]; }
        __syncthreads();
    }
    if (threadIdx.x == 0) { sums[2 * c] = s1[0]; sums[2 * c + 1] = s2[0]; }
}

__global__ void paconv_bn_apply_relu_kernel(float* __restrict__ buf, int rows, int ldc, int nch,
                                            const float* __restrict__ sums) {
    long i = (long)blockIdx.x * blockDim.x + threadIdx.x;
    if (i >= (long)rows * nch) return;
    int r = (int)(i / nch), c = (int)(i % nch);
    float mu = sums[2 * c] / (float)rows;
    float var = sums[2 * c + 1] / (float)rows - mu * mu;
    float v = (buf[(long)r * ldc + c] - mu) * rsqrtf(var + EPSBN);
    buf[(long)r * ldc + c] = v > 0.f ? v : 0.f;
}

/* ---------------- pooling: feat[b][c]=max_n, feat[b][1024+c]=mean_n ---------------- */
__global__ void paconv_pool_kernel(const float* __restrict__ out5, float* __restrict__ feat) {
    int t = blockIdx.x * blockDim.x + threadIdx.x;
    if (t >= ROWS) return;
    int b = t >> 10, c = t & 1023;
    float mx = -1e30f, s = 0.f;
    for (int n = 0; n < NPTS; ++n) {
        float v = out5[((long)(b << 10) + n) * 1024 + c];
        mx = fmaxf(mx, v); s += v;
    }
    feat[(long)b * 2048 + c] = mx;
    feat[(long)b * 2048 + 1024 + c] = s * (1.f / (float)NPTS);
}

/* ---------------- small dense layers ---------------- */
__global__ void paconv_linear_kernel(const float* __restrict__ A, const float* __restrict__ W,
                                     const float* __restrict__ bias, float* __restrict__ out,
                                     int Bn, int O, int Kk) {
    int t = blockIdx.x * blockDim.x + threadIdx.x;
    if (t >= Bn * O) return;
    int b = t / O, o = t % O;
    const float* a = A + (long)b * Kk;
    const float* w = W + (long)o * Kk;
    float s = 0.f;
    for (int k = 0; k < Kk; ++k) s += a[k] * w[k];
    if (bias) s += bias[o];
    out[t] = s;
}

__global__ void paconv_bn_rows_relu_kernel(float* __restrict__ z, int rowsB, int O) {
    int c = blockIdx.x * blockDim.x + threadIdx.x;
    if (c >= O) return;
    float s = 0.f, q = 0.f;
    for (int b = 0; b < rowsB; ++b) { float v = z[(long)b * O + c]; s += v; q += v * v; }
    float mu = s / (float)rowsB;
    float var = q / (float)rowsB - mu * mu;
    float inv = rsqrtf(var + EPSBN);
    for (int b = 0; b < rowsB; ++b) {
        float v = (z[(long)b * O + c] - mu) * inv;
        z[(long)b * O + c] = v > 0.f ? v : 0.f;
    }
}

/* =================================================================== */
extern "C" void kernel_launch(void* const* d_in, const int* in_sizes, int n_in,
                              void* d_out, int out_size, void* d_ws, size_t ws_size,
                              hipStream_t stream) {
    const float* x       = (const float*)d_in[0];
    const float* mats[4] = {(const float*)d_in[1], (const float*)d_in[2],
                            (const float*)d_in[3], (const float*)d_in[4]};
    const float* snw1[4] = {(const float*)d_in[5], (const float*)d_in[8],
                            (const float*)d_in[11], (const float*)d_in[14]};
    const float* snw2[4] = {(const float*)d_in[6], (const float*)d_in[9],
                            (const float*)d_in[12], (const float*)d_in[15]};
    const float* snb2[4] = {(const float*)d_in[7], (const float*)d_in[10],
                            (const float*)d_in[13], (const float*)d_in[16]};
    const float* conv5_w = (const float*)d_in[17];
    const float* lin1_w  = (const float*)d_in[18];
    const float* lin2_w  = (const float*)d_in[19];
    const float* lin3_w  = (const float*)d_in[20];
    const float* lin3_b  = (const float*)d_in[21];

    char* ws = (char*)d_ws;
    size_t off = 0;
    float* Point  = (float*)(ws + off); off += (size_t)ROWS * 2048 * 4;        /* 268 MB */
    float* Center = (float*)(ws + off); off += (size_t)ROWS * 2048 * 4;        /* 268 MB */
    float* Pcat   = (float*)(ws + off); off += (size_t)ROWS * 512 * 4;         /* 67 MB  */
    float* Out5   = (float*)(ws + off); off += (size_t)ROWS * 1024 * 4;        /* 134 MB */
    float* Score  = (float*)(ws + off); off += (size_t)ROWS * KNN * MSC * 4;   /* 21 MB  */
    int*   Idx    = (int*)  (ws + off); off += (size_t)ROWS * KNN * 4;         /* 2.6 MB */
    float* Xt     = (float*)(ws + off); off += (size_t)ROWS * 3 * 4;
    f16*   Ah     = (f16*)  (ws + off); off += (size_t)ROWS * 512 * 2;         /* 33.5 MB */
    f16*   Wp     = (f16*)  (ws + off); off += (size_t)2048 * 128 * 2;
    f16*   Wc     = (f16*)  (ws + off); off += (size_t)2048 * 128 * 2;
    f16*   W5h    = (f16*)  (ws + off); off += (size_t)1024 * 512 * 2;
    float* Stats  = (float*)(ws + off); off += (size_t)2048 * 2 * 4;
    float* SnSt   = (float*)(ws + off); off += 256;
    float* Feat   = (float*)(ws + off); off += (size_t)BATCH * 2048 * 4;
    float* Z1     = (float*)(ws + off); off += (size_t)BATCH * 512 * 4;
    float* Z2     = (float*)(ws + off); off += (size_t)BATCH * 256 * 4;

    /* ---- preprocessing ---- */
    paconv_xt_kernel<<<(ROWS * 3 + 255) / 256, 256, 0, stream>>>(x, Xt);
    paconv_knn_kernel<<<BATCH * (NPTS / 256), 256, 0, stream>>>(x, Idx);

    const int dL[4]     = {3, 64, 64, 128};
    const int KpadL[4]  = {32, 64, 64, 128};
    const int NcolL[4]  = {512, 512, 1024, 2048};
    const int oL[4]     = {64, 64, 128, 256};
    const int colOut[4] = {0, 64, 128, 256};
    const int colIn[4]  = {0, 0, 64, 128};   /* layer1 uses Xt */

    for (int l = 0; l < 4; ++l) {
        const int d = dL[l], Kpad = KpadL[l], Ncols = NcolL[l], o = oL[l];
        const int Mo = MSC * o;

        paconv_wprep_kernel<<<(Ncols * Kpad + 255) / 256, 256, 0, stream>>>(
            mats[l], d, Ncols, Kpad, Wp, Wc);

        hipMemsetAsync(SnSt, 0, 32 * sizeof(float), stream);
        paconv_sn_stats_kernel<<<ROWS, 320, 0, stream>>>(x, Idx, snw1[l], SnSt);
        paconv_sn_score_kernel<<<ROWS, 320, 0, stream>>>(x, Idx, snw1[l], snw2[l], snb2[l],
                                                         SnSt, Score);

        /* pack A to fragment-ready f16 */
        {
            const float* Asrc = (l == 0) ? Xt : (Pcat + colIn[l]);
            const int lda = (l == 0) ? 3 : 512;
            long total = (long)ROWS * Kpad;
            paconv_aprep_kernel<<<(int)((total + 255) / 256), 256, 0, stream>>>(
                Asrc, lda, d, Kpad, Ah, total);
        }

        const int waves = (ROWS / 16) * (Ncols / 64);
        const int gblocks = (waves + 7) / 8;
        paconv_gemm_wmma_kernel<<<gblocks, 256, 0, stream>>>(Ah, Wp, Kpad, Point, Mo, ROWS, Ncols);
        paconv_gemm_wmma_kernel<<<gblocks, 256, 0, stream>>>(Ah, Wc, Kpad, Center, Mo, ROWS, Ncols);

        paconv_assemble_kernel<<<ROWS, o, 0, stream>>>(Score, Idx, Point, Center, o,
                                                       Pcat + colOut[l], 512);
        paconv_bn_stats_kernel<<<o, 256, 0, stream>>>(Pcat + colOut[l], ROWS, 512, Stats);
        paconv_bn_apply_relu_kernel<<<(int)(((long)ROWS * o + 255) / 256), 256, 0, stream>>>(
            Pcat + colOut[l], ROWS, 512, o, Stats);
    }

    /* ---- conv5: (32768,512) x (512,1024) ---- */
    paconv_w5prep_kernel<<<(1024 * 512 + 255) / 256, 256, 0, stream>>>(conv5_w, W5h, 1024 * 512);
    {
        long total = (long)ROWS * 512;
        paconv_aprep_kernel<<<(int)((total + 255) / 256), 256, 0, stream>>>(
            Pcat, 512, 512, 512, Ah, total);
        const int waves = (ROWS / 16) * (1024 / 64);
        paconv_gemm_wmma_kernel<<<(waves + 7) / 8, 256, 0, stream>>>(Ah, W5h, 512,
                                                                     Out5, 1024, ROWS, 1024);
    }
    paconv_bn_stats_kernel<<<1024, 256, 0, stream>>>(Out5, ROWS, 1024, Stats);
    paconv_bn_apply_relu_kernel<<<(int)(((long)ROWS * 1024 + 255) / 256), 256, 0, stream>>>(
        Out5, ROWS, 1024, 1024, Stats);

    paconv_pool_kernel<<<(ROWS + 255) / 256, 256, 0, stream>>>(Out5, Feat);

    /* ---- FC head ---- */
    paconv_linear_kernel<<<(BATCH * 512 + 255) / 256, 256, 0, stream>>>(
        Feat, lin1_w, nullptr, Z1, BATCH, 512, 2048);
    paconv_bn_rows_relu_kernel<<<(512 + 255) / 256, 256, 0, stream>>>(Z1, BATCH, 512);

    paconv_linear_kernel<<<(BATCH * 256 + 255) / 256, 256, 0, stream>>>(
        Z1, lin2_w, nullptr, Z2, BATCH, 256, 512);
    paconv_bn_rows_relu_kernel<<<(256 + 255) / 256, 256, 0, stream>>>(Z2, BATCH, 256);

    paconv_linear_kernel<<<(BATCH * 40 + 255) / 256, 256, 0, stream>>>(
        Z2, lin3_w, lin3_b, (float*)d_out, BATCH, 40, 256);

    (void)in_sizes; (void)n_in; (void)out_size; (void)ws_size;
}
